// IMST_19713899888836
// MI455X (gfx1250) — matrix operand order repaired and checked by reading, not compiled, not run on hardware
//
#include <hip/hip_runtime.h>
#include <hip/hip_bf16.h>

// ---------------------------------------------------------------------------
// Segmented WCT on MI455X (gfx1250, wave32).
//  - features pre-converted to bf16 once; inner loops feed WMMA via
//    GLOBAL_LOAD_ASYNC_TO_LDS_B128 (ASYNCcnt) -- no VGPR round trip
//  - v_wmma_f32_16x16x32_bf16 for all pixel-dim GEMMs
//  - Newton-Schulz sqrt/invsqrt via split-bf16 (hi+lo) emulated-f32 GEMMs:
//    ~f32 accuracy at 8x the K-throughput of v_wmma_f32_16x16x4_f32
// Everything is L2-resident (192MB); compute-bound on the matrix pipe.
// Workspace: ~117 MB.
// ---------------------------------------------------------------------------

#define C_DIM 512
#define WH    16384
#define KC    8
#define KS    8
#define ALPHA 0.8f
#define EPS_F 1e-6f
#define NS_ITERS 12

typedef __attribute__((ext_vector_type(16))) __bf16 v16bf;
typedef __attribute__((ext_vector_type(8)))  float  v8f;

// ---- WMMA fragment builders (ISA 7.12.2 VGPR layouts) ----------------------
// A 16x32 bf16: lane L -> M = L%16; half = L/16: regs0-3 hold K=half*8+[0..7],
// regs4-7 hold K=16+half*8+[0..7], packed 2 per VGPR.
__device__ inline v16bf frag_a16(const __bf16* row32, int half) {
  v16bf f;
#pragma unroll
  for (int g = 0; g < 8; ++g) {
    int k0 = (g < 4 ? 2 * g : 16 + 2 * (g - 4)) + half * 8;
    f[2 * g]     = row32[k0];
    f[2 * g + 1] = row32[k0 + 1];
  }
  return f;
}
// B 32x16 bf16: lane L -> N = L%16; half = L/16: elems 0-7 hold K=half*8+[0..7],
// elems 8-15 hold K=16+half*8+[0..7] (same K order as A).
__device__ inline v16bf frag_b16(const __bf16* col32, int half) {
  v16bf f;
#pragma unroll
  for (int t = 0; t < 16; ++t) f[t] = col32[(t < 8 ? t : t + 8) + half * 8];
  return f;
}

// ---- CDNA5 async global->LDS copy (ASYNCcnt) -------------------------------
__device__ inline unsigned lds_off_of(const void* p) {
  return (unsigned)(uintptr_t)p;  // low 32 bits of flat LDS-aperture addr
}
__device__ inline void async_ld_b128(unsigned lds_off, const void* g) {
  asm volatile("global_load_async_to_lds_b128 %0, %1, off"
               :: "v"(lds_off), "v"((unsigned long long)(uintptr_t)g)
               : "memory");
}
__device__ inline void async_wait0() {
  asm volatile("s_wait_asynccnt 0x0" ::: "memory");
}

// ---- split-bf16: x ~= hi + lo, error ~2^-18 --------------------------------
__device__ inline void split_bf16(float x, __bf16& hi, __bf16& lo) {
  hi = (__bf16)x;
  lo = (__bf16)(x - (float)hi);
}

// ---------------------------------------------------------------------------
// Masked SYRK: out[k] = sum_{p: labels[p]==k} x_p x_p^T   (bf16 WMMA, f32 acc)
// grid (C/64, C/64, K); block 128 (4 waves). B tile streamed with async-to-LDS.
// ---------------------------------------------------------------------------
__global__ __launch_bounds__(128)
void k_masked_syrk(const __bf16* __restrict__ feat16, const int* __restrict__ labels,
                   float* __restrict__ out /* [K][C][C] */) {
  const int k    = blockIdx.z;
  const int row0 = blockIdx.x * 64;
  const int col0 = blockIdx.y * 64;
  __shared__ alignas(16) __bf16 sa[64][32];
  __shared__ alignas(16) __bf16 sb[64][32];
  const int tid  = threadIdx.x;
  const int wave = tid >> 5;
  const int lane = tid & 31;
  const int m    = lane & 15;
  const int half = lane >> 4;
  v8f acc[4] = {};
  for (int p0 = 0; p0 < WH; p0 += 32) {
    __syncthreads();
    // B tile: pure bf16 copy -> async DMA to LDS, 16B per issue
    for (int i = tid; i < 256; i += 128) {
      int r = i >> 2, c8 = (i & 3) * 8;
      async_ld_b128(lds_off_of(&sb[r][c8]),
                    feat16 + (size_t)(col0 + r) * WH + p0 + c8);
    }
    // A tile: masked select (cndmask, no converts)
    for (int i = tid; i < 64 * 32; i += 128) {
      int r = i >> 5, c = i & 31;
      int pix = p0 + c;
      __bf16 v = feat16[(size_t)(row0 + r) * WH + pix];
      sa[r][c] = (labels[pix] == k) ? v : (__bf16)0.0f;
    }
    async_wait0();
    __syncthreads();
    v16bf a = frag_a16(&sa[wave * 16 + m][0], half);
#pragma unroll
    for (int j = 0; j < 4; ++j) {
      v16bf b = frag_b16(&sb[j * 16 + m][0], half);
      acc[j] = __builtin_amdgcn_wmma_f32_16x16x32_bf16(
          false, a, false, b, (short)0, acc[j], false, false);
    }
  }
  float* o = out + (size_t)k * C_DIM * C_DIM;
#pragma unroll
  for (int j = 0; j < 4; ++j)
#pragma unroll
    for (int r = 0; r < 8; ++r) {
      int row = row0 + wave * 16 + half * 8 + r;
      int col = col0 + j * 16 + m;
      o[(size_t)row * C_DIM + col] = acc[j][r];
    }
}

// ---------------------------------------------------------------------------
// Batched 512^3 emulated-f32 GEMM via split-bf16:
//   D[b] = alpha*A[b]@B[b] + diagAdd*I,  A@B ~= Ah@Bh + Ah@Bl + Al@Bh
// grid (8,8,nbat); block 128 (4 waves); 64x64 tile; K-chunk 32.
// ---------------------------------------------------------------------------
__global__ __launch_bounds__(128)
void k_gemm512_split(const float* __restrict__ A, const float* __restrict__ B,
                     float* __restrict__ D, float alpha, float diagAdd) {
  const size_t PER = (size_t)C_DIM * C_DIM;
  const int row0 = blockIdx.x * 64;
  const int col0 = blockIdx.y * 64;
  const float* Ab = A + (size_t)blockIdx.z * PER;
  const float* Bb = B + (size_t)blockIdx.z * PER;
  float*       Db = D + (size_t)blockIdx.z * PER;
  __shared__ alignas(16) __bf16 sah[64][32], sal[64][32];
  __shared__ alignas(16) __bf16 sbh[64][32], sbl[64][32];  // [n][k]
  const int tid = threadIdx.x, wave = tid >> 5, lane = tid & 31;
  const int m = lane & 15, half = lane >> 4;
  v8f acc[4] = {};
  for (int k0 = 0; k0 < C_DIM; k0 += 32) {
    __syncthreads();
    for (int i = tid; i < 64 * 32; i += 128) {  // A tile, coalesced in k
      int r = i >> 5, c = i & 31;
      __bf16 hi, lo;
      split_bf16(Ab[(size_t)(row0 + r) * C_DIM + k0 + c], hi, lo);
      sah[r][c] = hi; sal[r][c] = lo;
    }
    for (int i = tid; i < 64 * 32; i += 128) {  // B tile, coalesced in n
      int kk = i >> 6, n = i & 63;
      __bf16 hi, lo;
      split_bf16(Bb[(size_t)(k0 + kk) * C_DIM + col0 + n], hi, lo);
      sbh[n][kk] = hi; sbl[n][kk] = lo;
    }
    __syncthreads();
    v16bf ah = frag_a16(&sah[wave * 16 + m][0], half);
    v16bf al = frag_a16(&sal[wave * 16 + m][0], half);
#pragma unroll
    for (int j = 0; j < 4; ++j) {
      v16bf bh = frag_b16(&sbh[j * 16 + m][0], half);
      v16bf bl = frag_b16(&sbl[j * 16 + m][0], half);
      acc[j] = __builtin_amdgcn_wmma_f32_16x16x32_bf16(false, ah, false, bh, (short)0, acc[j], false, false);
      acc[j] = __builtin_amdgcn_wmma_f32_16x16x32_bf16(false, ah, false, bl, (short)0, acc[j], false, false);
      acc[j] = __builtin_amdgcn_wmma_f32_16x16x32_bf16(false, al, false, bh, (short)0, acc[j], false, false);
    }
  }
#pragma unroll
  for (int j = 0; j < 4; ++j)
#pragma unroll
    for (int r = 0; r < 8; ++r) {
      int row = row0 + wave * 16 + half * 8 + r;
      int col = col0 + j * 16 + m;
      float v = alpha * acc[j][r];
      if (row == col) v += diagAdd;
      Db[(size_t)row * C_DIM + col] = v;
    }
}

// ---------------------------------------------------------------------------
// Apply: out[:,p] = Malpha[k]@cf[:,p] + bias[k] + (1-ALPHA)*cf[:,p], k=label[p]
// Disjoint masks -> predicate stores only (WMMA runs with full EXEC).
// grid (C/64, WH/128, KC); block 128 (4 waves); M tile via async-to-LDS.
// ---------------------------------------------------------------------------
__global__ __launch_bounds__(128)
void k_apply(const __bf16* __restrict__ cf16, const float* __restrict__ cf,
             const int* __restrict__ labels,
             const __bf16* __restrict__ M16 /*[KC][C][C], alpha folded*/,
             const float* __restrict__ bias /*[KC][C]*/,
             float* __restrict__ out) {
  const int kc   = blockIdx.z;
  const int row0 = blockIdx.x * 64;
  const int pix0 = blockIdx.y * 128;
  __shared__ alignas(16) __bf16 sm[64][32];
  __shared__ alignas(16) __bf16 sx[128][32];
  const int tid = threadIdx.x, wave = tid >> 5, lane = tid & 31;
  const int m = lane & 15, half = lane >> 4;
  v8f acc[8] = {};
  for (int d0 = 0; d0 < C_DIM; d0 += 32) {
    __syncthreads();
    // M tile: async DMA (pure copy)
    for (int i = tid; i < 256; i += 128) {
      int r = i >> 2, c8 = (i & 3) * 8;
      async_ld_b128(lds_off_of(&sm[r][c8]),
                    M16 + ((size_t)kc * C_DIM + row0 + r) * C_DIM + d0 + c8);
    }
    // cf tile: transposed into [pixel][k]; vectorized global reads
    for (int i = tid; i < 512; i += 128) {
      int kk = i >> 4, p8 = (i & 15) * 8;
      const __bf16* src = cf16 + (size_t)(d0 + kk) * WH + pix0 + p8;
#pragma unroll
      for (int t = 0; t < 8; ++t) sx[p8 + t][kk] = src[t];
    }
    async_wait0();
    __syncthreads();
    v16bf a = frag_a16(&sm[wave * 16 + m][0], half);
#pragma unroll
    for (int j = 0; j < 8; ++j) {
      v16bf b = frag_b16(&sx[j * 16 + m][0], half);
      acc[j] = __builtin_amdgcn_wmma_f32_16x16x32_bf16(
          false, a, false, b, (short)0, acc[j], false, false);
    }
  }
#pragma unroll
  for (int j = 0; j < 8; ++j) {
    int pixel = pix0 + j * 16 + m;
    if (labels[pixel] == kc) {
#pragma unroll
      for (int r = 0; r < 8; ++r) {
        int row = row0 + wave * 16 + half * 8 + r;
        out[(size_t)row * WH + pixel] =
            acc[j][r] + bias[(size_t)kc * C_DIM + row]
            + (1.0f - ALPHA) * cf[(size_t)row * WH + pixel];
      }
    }
  }
}

// ---------------------------------------------------------------------------
// Small helper kernels
// ---------------------------------------------------------------------------
__global__ void k_zero(float* p, int n) {
  int i = blockIdx.x * blockDim.x + threadIdx.x;
  if (i < n) p[i] = 0.f;
}

__global__ void k_cvt_bf16(const float* __restrict__ s, __bf16* __restrict__ d, int n) {
  int i = blockIdx.x * 256 + threadIdx.x;
  if (i < n) d[i] = (__bf16)s[i];
}

__global__ void k_counts(const int* labels, float* counts, int n) {
  int i = blockIdx.x * blockDim.x + threadIdx.x;
  if (i < n) atomicAdd(&counts[labels[i]], 1.0f);  // integer-valued: exact
}

// Deterministic per-(cluster,channel) sums: one block per channel, tree reduce.
__global__ __launch_bounds__(128)
void k_label_sums(const float* __restrict__ feat, const int* __restrict__ labels,
                  float* __restrict__ sums /*[K][C]*/, int wh) {
  __shared__ float red[KC][128];
  int c = blockIdx.x;
  float acc[KC] = {};
  for (int p = threadIdx.x; p < wh; p += 128) {
    float v = feat[(size_t)c * wh + p];
    int l = labels[p];
#pragma unroll
    for (int k = 0; k < KC; ++k) acc[k] += (l == k) ? v : 0.f;
  }
#pragma unroll
  for (int k = 0; k < KC; ++k) red[k][threadIdx.x] = acc[k];
  __syncthreads();
  for (int s = 64; s > 0; s >>= 1) {
    if (threadIdx.x < s)
#pragma unroll
      for (int k = 0; k < KC; ++k) red[k][threadIdx.x] += red[k][threadIdx.x + s];
    __syncthreads();
  }
  if (threadIdx.x < KC) sums[(size_t)threadIdx.x * C_DIM + c] = red[threadIdx.x][0];
}

// Weighted style mean per content cluster: grid KC, block 512.
__global__ __launch_bounds__(512)
void k_style_stats(const float* scores, const float* ssums, const float* scounts,
                   float* smean_w, float* totals) {
  int kc = blockIdx.x, c = threadIdx.x;
  float tot = 0.f, mn = 0.f;
#pragma unroll
  for (int ks = 0; ks < KS; ++ks) {
    float w = scores[kc * KS + ks];
    tot += w * scounts[ks];
    mn  += w * ssums[ks * C_DIM + c];
  }
  if (c == 0) totals[kc] = tot;
  smean_w[kc * C_DIM + c] = mn / tot;
}

// Covariances from raw second moments (mean folded analytically).
__global__ __launch_bounds__(256)
void k_assemble(const float* __restrict__ M2, const float* __restrict__ csums,
                const float* __restrict__ ccounts, const float* __restrict__ scores,
                const float* __restrict__ smean_w, const float* __restrict__ totals,
                float* __restrict__ COV) {
  size_t idx = (size_t)blockIdx.x * 256 + threadIdx.x;
  size_t per = (size_t)C_DIM * C_DIM;
  int mat = (int)(idx / per);
  size_t rc = idx % per;
  int r = (int)(rc / C_DIM), c = (int)(rc % C_DIM);
  float v;
  if (mat < KC) {
    float n  = ccounts[mat];
    float mr = csums[mat * C_DIM + r] / n;
    float mc = csums[mat * C_DIM + c] / n;
    v = (M2[idx] - n * mr * mc) / (n - 1.0f);
    if (r == c) v += EPS_F;  // (e+eps)^-1/2 == invsqrt(cov + eps I)
  } else {
    int kc = mat - KC;
    float acc = 0.f;
#pragma unroll
    for (int ks = 0; ks < KS; ++ks)
      acc += scores[kc * KS + ks] * M2[((size_t)(KC + ks) * C_DIM + r) * C_DIM + c];
    float tot = totals[kc];
    v = (acc - tot * smean_w[kc * C_DIM + r] * smean_w[kc * C_DIM + c]) / (tot - 1.0f);
  }
  COV[idx] = v;
}

// Frobenius norm^2, deterministic two-stage reduce.
__global__ __launch_bounds__(256)
void k_fro_part(const float* __restrict__ A, float* __restrict__ part) {
  __shared__ float red[256];
  size_t idx = (size_t)blockIdx.x * 256 + threadIdx.x;
  float v = A[idx];
  red[threadIdx.x] = v * v;
  __syncthreads();
  for (int s = 128; s > 0; s >>= 1) {
    if (threadIdx.x < s) red[threadIdx.x] += red[threadIdx.x + s];
    __syncthreads();
  }
  if (threadIdx.x == 0) part[blockIdx.x] = red[0];
}
__global__ __launch_bounds__(256)
void k_fro_final(const float* __restrict__ part, float* __restrict__ fro) {
  __shared__ float red[256];
  float a = 0.f;
  for (int i = threadIdx.x; i < 1024; i += 256) a += part[blockIdx.x * 1024 + i];
  red[threadIdx.x] = a;
  __syncthreads();
  for (int s = 128; s > 0; s >>= 1) {
    if (threadIdx.x < s) red[threadIdx.x] += red[threadIdx.x + s];
    __syncthreads();
  }
  if (threadIdx.x == 0) fro[blockIdx.x] = red[0];
}

// Normalize Y0 = cov/||cov||_F in place, Z0 = I.
__global__ __launch_bounds__(256)
void k_ns_init(float* __restrict__ Y, float* __restrict__ Z, const float* __restrict__ fro) {
  size_t idx = (size_t)blockIdx.x * 256 + threadIdx.x;
  size_t per = (size_t)C_DIM * C_DIM;
  int mat = (int)(idx / per);
  size_t rc = idx % per;
  float s = sqrtf(fro[mat]) + 1e-20f;
  Y[idx] = Y[idx] / s;
  Z[idx] = ((rc / C_DIM) == (rc % C_DIM)) ? 1.0f : 0.0f;
}

// Un-normalize: Wc = Z_content/sqrt(s_c) (cov^-1/2); Cs = Y_style*sqrt(s_s) (cov^1/2)
__global__ __launch_bounds__(256)
void k_pack_WC(const float* __restrict__ Yf, const float* __restrict__ Zf,
               const float* __restrict__ fro, float* __restrict__ Wc, float* __restrict__ Cs) {
  size_t idx = (size_t)blockIdx.x * 256 + threadIdx.x;  // over 8*C*C
  size_t per = (size_t)C_DIM * C_DIM;
  int k = (int)(idx / per);
  size_t rc = idx % per;
  float sc = sqrtf(fro[k]) + 1e-20f;
  float ss = sqrtf(fro[KC + k]) + 1e-20f;
  Wc[idx] = Zf[(size_t)k * per + rc] / sqrtf(sc);
  Cs[idx] = Yf[(size_t)(KC + k) * per + rc] * sqrtf(ss);
}

// bias[k] = ALPHA*smean_w[k] - Malpha[k] @ mu_k  (Malpha has ALPHA folded)
__global__ __launch_bounds__(256)
void k_bias(const float* __restrict__ Mm, const float* __restrict__ smean_w,
            const float* __restrict__ csums, const float* __restrict__ ccounts,
            float* __restrict__ bias) {
  int k = blockIdx.y;
  int c = blockIdx.x * 256 + threadIdx.x;
  float n = ccounts[k];
  float acc = 0.f;
  for (int d = 0; d < C_DIM; ++d)
    acc += Mm[((size_t)k * C_DIM + c) * C_DIM + d] * (csums[k * C_DIM + d] / n);
  bias[k * C_DIM + c] = ALPHA * smean_w[k * C_DIM + c] - acc;
}

// ---------------------------------------------------------------------------
extern "C" void kernel_launch(void* const* d_in, const int* in_sizes, int n_in,
                              void* d_out, int out_size, void* d_ws, size_t ws_size,
                              hipStream_t stream) {
  (void)in_sizes; (void)n_in; (void)out_size; (void)ws_size;
  const float* cf     = (const float*)d_in[0];
  const float* sf     = (const float*)d_in[1];
  const float* scores = (const float*)d_in[2];
  const int*   cl     = (const int*)d_in[3];
  const int*   sl     = (const int*)d_in[4];
  float*       out    = (float*)d_out;

  char* ws = (char*)d_ws;
  const size_t PER  = (size_t)C_DIM * C_DIM;     // 262144
  const size_t MATS = 16 * PER * sizeof(float);  // 16 MB (== C_DIM*WH*2 bytes)
  float*  M2   = (float*)(ws + 0 * MATS);  // [16][C][C]; then T; then Malpha
  float*  Y0   = (float*)(ws + 1 * MATS);  // cov -> normalized -> NS Y
  float*  Z0   = (float*)(ws + 2 * MATS);
  float*  Y1   = (float*)(ws + 3 * MATS);  // NS ping; later Wc
  float*  Z1   = (float*)(ws + 4 * MATS);
  __bf16* CF16 = (__bf16*)(ws + 5 * MATS); // 512x16384 bf16 (16MB)
  __bf16* SF16 = (__bf16*)(ws + 6 * MATS);
  __bf16* M16  = (__bf16*)(ws + 7 * MATS); // 8 x 512x512 bf16 (4MB)
  float*  S    = (float*)(ws + 7 * MATS + 8 * PER * sizeof(__bf16));
  float* csums   = S;            // 4096
  float* ssums   = S + 4096;     // 4096
  float* ccounts = S + 8192;     // 8
  float* scounts = S + 8200;     // 8
  float* smean_w = S + 8208;     // 4096
  float* totals  = S + 12304;    // 8
  float* fro     = S + 12312;    // 16
  float* bias    = S + 12328;    // 4096
  float* part    = S + 16424;    // 16*1024 partials
  float* T  = M2;
  float* Mf = M2;   // Malpha f32 lives in first 8MB of M2 after T is dead
  float* Wc = Y1;
  float* Cs = Z1;

  // 1) zero stat scratch
  k_zero<<<dim3((12328 + 255) / 256), dim3(256), 0, stream>>>(S, 12328);
  // 2) per-cluster counts and channel sums (f32 inputs, deterministic)
  k_counts<<<dim3(WH / 256), dim3(256), 0, stream>>>(cl, ccounts, WH);
  k_counts<<<dim3(WH / 256), dim3(256), 0, stream>>>(sl, scounts, WH);
  k_label_sums<<<dim3(C_DIM), dim3(128), 0, stream>>>(cf, cl, csums, WH);
  k_label_sums<<<dim3(C_DIM), dim3(128), 0, stream>>>(sf, sl, ssums, WH);
  // 3) one-shot bf16 conversion of the feature maps
  k_cvt_bf16<<<dim3(C_DIM * WH / 256), dim3(256), 0, stream>>>(cf, CF16, C_DIM * WH);
  k_cvt_bf16<<<dim3(C_DIM * WH / 256), dim3(256), 0, stream>>>(sf, SF16, C_DIM * WH);
  // 4) masked second moments (bf16 WMMA; async-to-LDS B tiles)
  dim3 gs(C_DIM / 64, C_DIM / 64, KC);
  k_masked_syrk<<<gs, dim3(128), 0, stream>>>(CF16, cl, M2);            // slots 0..7
  k_masked_syrk<<<gs, dim3(128), 0, stream>>>(SF16, sl, M2 + 8 * PER);  // slots 8..15
  // 5) weighted style means / totals
  k_style_stats<<<dim3(KC), dim3(C_DIM), 0, stream>>>(scores, ssums, scounts, smean_w, totals);
  // 6) assemble 16 covariances into Y0 (content +EPS*I)
  k_assemble<<<dim3((unsigned)(16 * PER / 256)), dim3(256), 0, stream>>>(
      M2, csums, ccounts, scores, smean_w, totals, Y0);
  // 7) Frobenius norms (deterministic two-stage)
  k_fro_part<<<dim3((unsigned)(16 * PER / 256)), dim3(256), 0, stream>>>(Y0, part);
  k_fro_final<<<dim3(16), dim3(256), 0, stream>>>(part, fro);
  // 8) normalize + init Z = I
  k_ns_init<<<dim3((unsigned)(16 * PER / 256)), dim3(256), 0, stream>>>(Y0, Z0, fro);
  // 9) coupled Newton-Schulz: T = 1.5I - 0.5 Z@Y; Y' = Y@T; Z' = T@Z
  //    (split-bf16 emulated-f32 GEMMs on the bf16 matrix pipe)
  float* Ya = Y0; float* Za = Z0; float* Yb = Y1; float* Zb = Z1;
  dim3 gg(C_DIM / 64, C_DIM / 64, 16);
  for (int it = 0; it < NS_ITERS; ++it) {
    k_gemm512_split<<<gg, dim3(128), 0, stream>>>(Za, Ya, T, -0.5f, 1.5f);
    k_gemm512_split<<<gg, dim3(128), 0, stream>>>(Ya, T, Yb, 1.0f, 0.0f);
    k_gemm512_split<<<gg, dim3(128), 0, stream>>>(T, Za, Zb, 1.0f, 0.0f);
    float* t;
    t = Ya; Ya = Yb; Yb = t;
    t = Za; Za = Zb; Zb = t;
  }
  // NS_ITERS even -> results in Y0/Z0; Y1/Z1 free for Wc/Cs
  // 10) un-normalize roots
  k_pack_WC<<<dim3((unsigned)(8 * PER / 256)), dim3(256), 0, stream>>>(Ya, Za, fro, Wc, Cs);
  // 11) Malpha = ALPHA * Cs @ Wc  (T dead, reuse M2)
  dim3 gm(C_DIM / 64, C_DIM / 64, KC);
  k_gemm512_split<<<gm, dim3(128), 0, stream>>>(Cs, Wc, Mf, ALPHA, 0.0f);
  // 12) bias = ALPHA*smean - Malpha @ mu; bf16 copy of Malpha for the apply GEMM
  k_bias<<<dim3(2, KC), dim3(256), 0, stream>>>(Mf, smean_w, csums, ccounts, bias);
  k_cvt_bf16<<<dim3((unsigned)(8 * PER / 256)), dim3(256), 0, stream>>>(Mf, M16, (int)(8 * PER));
  // 13) fused apply + scatter (bf16 WMMA; stores predicated by label)
  dim3 ga(C_DIM / 64, WH / 128, KC);
  k_apply<<<ga, dim3(128), 0, stream>>>(CF16, cf, cl, M16, bias, out);
}
